// PQMatcher_77223511982260
// MI455X (gfx1250) — matrix operand order repaired and checked by reading, not compiled, not run on hardware
//
#include <hip/hip_runtime.h>
#include <hip/hip_bf16.h>
#include <cmath>

// ---------------------------------------------------------------------------
// MI455X (gfx1250) implementation of the attention+GRU scan.
//   Phase A (parallel, WMMA): Wup[t] = Up@Wpc^T for all t; Wuq = Uq@Wqc^T.
//   Phase B (sequential, one persistent 1024-thread workgroup, 32 wave32
//            waves, 284KB LDS): 500 steps of attention + gated residual +
//            GRU, with all per-step GEMMs done via v_wmma_f32_16x16x32_bf16.
// ---------------------------------------------------------------------------

#define LP 500
#define LQ 64
#define BB 64
#define DD 256
#define HH 150
#define HP 160   // H padded to multiple of 16

typedef __attribute__((ext_vector_type(16))) __bf16 v16bf;
typedef __attribute__((ext_vector_type(8)))  __bf16 v8bf;
typedef __attribute__((ext_vector_type(8)))  float  v8f;

__device__ __forceinline__ v8f wmma_bf16(v16bf a, v16bf b, v8f c) {
  return __builtin_amdgcn_wmma_f32_16x16x32_bf16(false, a, false, b,
                                                 (short)0, c, false, false);
}

// A-fragment (16x32 bf16): lane m=lane&15 holds row M=m; half=lane>>4 selects
// K-chunks {half*8 .. +7} and {16+half*8 .. +7} (ISA 7.12.2).
__device__ __forceinline__ v16bf make_frag(const __bf16* c0, const __bf16* c1) {
  v8bf a = *(const v8bf*)c0;
  v8bf b = *(const v8bf*)c1;
  v16bf r;
#pragma unroll
  for (int i = 0; i < 8; ++i) { r[i] = a[i]; r[8 + i] = b[i]; }
  return r;
}

// base points at [row0][k0=0] of the A tile (row stride ld elements).
__device__ __forceinline__ v16bf load_a_frag(const __bf16* base, int ld,
                                             int k0, int lane) {
  int m = lane & 15, half = lane >> 4;
  const __bf16* p = base + (size_t)m * ld + k0 + half * 8;
  return make_frag(p, p + 16);
}

// B-fragment (32x16 bf16) from row-major weights W[N][K]: lane col=lane&15,
// khalf=lane>>4 -> K-chunks {khalf*8 .. +7} and {16+khalf*8 .. +7}.
__device__ __forceinline__ v16bf load_b_frag(const __bf16* W, int ldk,
                                             int nbase, int k0, int lane) {
  int col = lane & 15, khalf = lane >> 4;
  const __bf16* p = W + (size_t)(nbase + col) * ldk + k0 + khalf * 8;
  return make_frag(p, p + 16);
}

__device__ __forceinline__ float sigmoidf_(float x) {
  return 1.0f / (1.0f + __expf(-x));
}

// ------------------------- prep kernels (VALU) -----------------------------

__global__ void k_f32_to_bf16(const float* __restrict__ s,
                              __bf16* __restrict__ d, int n) {
  int i = blockIdx.x * blockDim.x + threadIdx.x;
  if (i < n) d[i] = (__bf16)s[i];
}

// Wc[160][256] = W[:,:256]+W[:,256:] (rows>=150 zero).  W is [150][512].
__global__ void k_build_wc(const float* __restrict__ W, __bf16* __restrict__ d) {
  int i = blockIdx.x * blockDim.x + threadIdx.x;
  if (i >= HP * DD) return;
  int n = i / DD, k = i % DD;
  float v = (n < HH) ? (W[n * 2 * DD + k] + W[n * 2 * DD + DD + k]) : 0.0f;
  d[i] = (__bf16)v;
}

// Wgc[512][1024] = rows 512..1023 of Wg.
__global__ void k_build_wg(const float* __restrict__ Wg, __bf16* __restrict__ d) {
  int i = blockIdx.x * blockDim.x + threadIdx.x;
  if (i >= 512 * 1024) return;
  int n = i / 1024, k = i % 1024;
  d[i] = (__bf16)Wg[(size_t)(512 + n) * 1024 + k];
}

// Wihb[480][512]: gate-padded rows (3 blocks of 160, rows 150..159 zero).
__global__ void k_build_wih(const float* __restrict__ W, __bf16* __restrict__ d) {
  int i = blockIdx.x * blockDim.x + threadIdx.x;
  if (i >= 3 * HP * 512) return;
  int n = i / 512, k = i % 512;
  int g = n / HP, h = n % HP;
  d[i] = (__bf16)((h < HH) ? W[(size_t)(g * HH + h) * 512 + k] : 0.0f);
}

// Whhb[480][160]: gate-padded rows, K padded 150->160.
__global__ void k_build_whh(const float* __restrict__ W, __bf16* __restrict__ d) {
  int i = blockIdx.x * blockDim.x + threadIdx.x;
  if (i >= 3 * HP * HP) return;
  int n = i / HP, k = i % HP;
  int g = n / HP, h = n % HP;
  d[i] = (__bf16)((h < HH && k < HH) ? W[(size_t)(g * HH + h) * HH + k] : 0.0f);
}

// Wvb[160][160]
__global__ void k_build_wv(const float* __restrict__ W, __bf16* __restrict__ d) {
  int i = blockIdx.x * blockDim.x + threadIdx.x;
  if (i >= HP * HP) return;
  int n = i / HP, k = i % HP;
  d[i] = (__bf16)((n < HH && k < HH) ? W[n * HH + k] : 0.0f);
}

// bias pad [450] -> [480]
__global__ void k_build_bias(const float* __restrict__ b, float* __restrict__ d) {
  int i = blockIdx.x * blockDim.x + threadIdx.x;
  if (i >= 3 * HP) return;
  int g = i / HP, h = i % HP;
  d[i] = (h < HH) ? b[g * HH + h] : 0.0f;
}

// -------------------- parallel WMMA GEMM (hoisted work) --------------------
// out[M x 160] (f32) = A[M x 256] (bf16) @ Wc[160 x 256]^T.  One 16x16 tile
// per wave; K = 256 -> 8 WMMAs.
__global__ void k_proj_gemm(const __bf16* __restrict__ A,
                            const __bf16* __restrict__ Wc,
                            float* __restrict__ out, int mtiles) {
  const int lane = threadIdx.x & 31;
  const int wave = threadIdx.x >> 5;
  const int wpb  = blockDim.x >> 5;
  int tile = blockIdx.x * wpb + wave;
  int tiles = mtiles * (HP / 16);
  if (tile >= tiles) return;
  int tm = tile / (HP / 16);
  int tn = tile % (HP / 16);
  v8f acc = {};
  const __bf16* arow = A + (size_t)tm * 16 * DD;
#pragma unroll
  for (int k0 = 0; k0 < DD; k0 += 32) {
    v16bf af = load_a_frag(arow, DD, k0, lane);
    v16bf bf_ = load_b_frag(Wc, DD, tn * 16, k0, lane);
    acc = wmma_bf16(af, bf_, acc);
  }
  int col = tn * 16 + (lane & 15);
  int rb  = tm * 16 + (lane >> 4) * 8;
#pragma unroll
  for (int r = 0; r < 8; ++r)
    out[(size_t)(rb + r) * HP + col] = acc[r];
}

// ----------------------- persistent sequential scan ------------------------
// LDS layout (bytes):
//   [0      , 65536 ) cbuf  : bf16 [64][512]  attention context (dup), then c_
//   [65536  , 86016 ) vbuf  : bf16 [64][160]  GRU state (padded, bf16)
//   [86016  , 126976) vf32  : f32  [64][160]  GRU state (f32)
//   [126976 , 290816) union :
//        attention:  vWv f32 [64][160] (40960) | s/a f32 [64][64] (16384)
//        gates    :  grz f32 [64][320] (81920) | i_n f32 [64][160] | h_n f32 [64][160]
#define SMEM_BYTES 290816

__global__ void __launch_bounds__(1024, 1)
k_seq_scan(const __bf16* __restrict__ Upbf,     // [500*64][256]
           const float*  __restrict__ WupAll,   // [500*64][160]
           const float*  __restrict__ WuqAll,   // [64*64][160]
           const float*  __restrict__ Uq,       // [64*64][256] f32 (input)
           const float*  __restrict__ V,        // [64][150]
           const __bf16* __restrict__ Wgc,      // [512][1024]
           const __bf16* __restrict__ Wihb,     // [480][512]
           const __bf16* __restrict__ Whhb,     // [480][160]
           const __bf16* __restrict__ Wvb,      // [160][160]
           const float*  __restrict__ bihp,     // [480]
           const float*  __restrict__ bhhp,     // [480]
           const float*  __restrict__ v0,       // [64][150]
           float* __restrict__ out) {           // [500][64][150]
  extern __shared__ char smem[];
  __bf16* cbuf = (__bf16*)smem;
  __bf16* vbuf = (__bf16*)(smem + 65536);
  float*  vf32 = (float*)(smem + 86016);
  char*   un   = smem + 126976;
  float* vWv  = (float*)un;
  float* sbuf = (float*)(un + 40960);
  float* grz  = (float*)un;
  float* inb  = (float*)(un + 81920);
  float* hnb  = (float*)(un + 122880);

  const int tid  = threadIdx.x;
  const int lane = tid & 31;
  const int wave = tid >> 5;
  const int m    = lane & 15;
  const int half = lane >> 4;

  // init GRU state
  for (int p = tid; p < BB * HP; p += 1024) {
    int b = p / HP, h = p % HP;
    float v = (h < HH) ? v0[b * HH + h] : 0.0f;
    vf32[p] = v;
    vbuf[p] = (__bf16)v;
  }
  __syncthreads();

  for (int t = 0; t < LP; ++t) {
    // ---- phase 1: vWv = v @ Wv^T  (WMMA, 4x10 tiles, K=160) ----
    for (int tile = wave; tile < 4 * (HP / 16); tile += 32) {
      int tm = tile / (HP / 16), tn = tile % (HP / 16);
      v8f acc = {};
      const __bf16* vrow = vbuf + (size_t)tm * 16 * HP;
#pragma unroll
      for (int k0 = 0; k0 < HP; k0 += 32) {
        v16bf af = load_a_frag(vrow, HP, k0, lane);
        v16bf bf_ = load_b_frag(Wvb, HP, tn * 16, k0, lane);
        acc = wmma_bf16(af, bf_, acc);
      }
      int col = tn * 16 + m;
      int rb  = tm * 16 + half * 8;
#pragma unroll
      for (int r = 0; r < 8; ++r) vWv[(rb + r) * HP + col] = acc[r];
    }
    // prefetch next step's A operand while we have memory slack
    if (t + 1 < LP && tid == 0)
      __builtin_prefetch(Upbf + (size_t)(t + 1) * BB * DD, 0, 1);
    __syncthreads();

    // ---- phase 2: scores s[b][q] = sum_h V[b,h]*tanh(Wup+Wuq+vWv) ----
    for (int p = tid; p < BB * LQ; p += 1024) {
      int q = p >> 6, b = p & 63;
      const float* wup = WupAll + ((size_t)t * BB + b) * HP;
      const float* wuq = WuqAll + ((size_t)q * BB + b) * HP;
      const float* vv  = vWv + b * HP;
      const float* Vb  = V + b * HH;
      float sum = 0.0f;
      for (int h = 0; h < HH; ++h)
        sum += Vb[h] * tanhf(wup[h] + wuq[h] + vv[h]);
      sbuf[b * LQ + q] = sum;
    }
    __syncthreads();
    // softmax over q (one row per thread)
    if (tid < BB) {
      float mx = -1e30f;
      for (int q = 0; q < LQ; ++q) mx = fmaxf(mx, sbuf[tid * LQ + q]);
      float sum = 0.0f;
      for (int q = 0; q < LQ; ++q) {
        float e = __expf(sbuf[tid * LQ + q] - mx);
        sbuf[tid * LQ + q] = e;
        sum += e;
      }
      float inv = 1.0f / sum;
      for (int q = 0; q < LQ; ++q) sbuf[tid * LQ + q] *= inv;
    }
    __syncthreads();

    // ---- phase 3: context c[b][d] = sum_q a[b,q]*Uq[q,b,d]; dup to 512 ----
    for (int p = tid; p < BB * DD; p += 1024) {
      int b = p >> 8, d = p & 255;
      const float* arow = sbuf + b * LQ;
      float sum = 0.0f;
      for (int q = 0; q < LQ; ++q)
        sum += arow[q] * Uq[((size_t)q * BB + b) * DD + d];
      __bf16 cv = (__bf16)sum;
      cbuf[b * 512 + d]       = cv;
      cbuf[b * 512 + DD + d]  = cv;
    }
    __syncthreads();

    // ---- phase 4: g = sigmoid(r @ Wg[512:]^T); c_ = g * c2 (in place) ----
    // r columns: [0,256)=Up_t, [256,512)=Up_t, [512,1024)=cbuf.  K=1024.
    {
      int tm  = wave >> 3;            // 0..3
      int tnb = (wave & 7) * 4;       // 4 N-tiles per wave, 32 total
      v8f acc4[4] = {v8f{}, v8f{}, v8f{}, v8f{}};
      const __bf16* uprow = Upbf + ((size_t)t * BB + tm * 16 + m) * DD;
      const __bf16* crow  = cbuf + (size_t)(tm * 16 + m) * 512;
#pragma unroll 8
      for (int k0 = 0; k0 < 1024; k0 += 32) {
        int cb = k0 + half * 8;
        const __bf16* p0 = (cb < 512) ? (uprow + (cb & 255)) : (crow + cb - 512);
        int cb1 = cb + 16;
        const __bf16* p1 = (cb1 < 512) ? (uprow + (cb1 & 255)) : (crow + cb1 - 512);
        v16bf af = make_frag(p0, p1);
#pragma unroll
        for (int i = 0; i < 4; ++i) {
          v16bf bf_ = load_b_frag(Wgc, 1024, (tnb + i) * 16, k0, lane);
          acc4[i] = wmma_bf16(af, bf_, acc4[i]);
        }
      }
      __syncthreads();   // all reads of cbuf done before in-place update
#pragma unroll
      for (int i = 0; i < 4; ++i) {
        int nb = (tnb + i) * 16 + m;
#pragma unroll
        for (int r = 0; r < 8; ++r) {
          int rr = tm * 16 + half * 8 + r;
          float c2 = (float)cbuf[rr * 512 + nb];
          cbuf[rr * 512 + nb] = (__bf16)(sigmoidf_(acc4[i][r]) * c2);
        }
      }
    }
    __syncthreads();

    // ---- phase 5: gate GEMMs gi = c_ @ W_ih^T, gh = v @ W_hh^T ----
    for (int t5 = wave; t5 < 4 * 30; t5 += 32) {
      int tm = t5 / 30, tn = t5 % 30;
      v8f ai = {}, ah = {};
      const __bf16* ctile = cbuf + (size_t)tm * 16 * 512;   // tile base (no +m)
#pragma unroll 8
      for (int k0 = 0; k0 < 512; k0 += 32) {
        v16bf af = load_a_frag(ctile, 512, k0, lane);
        v16bf bf_ = load_b_frag(Wihb, 512, tn * 16, k0, lane);
        ai = wmma_bf16(af, bf_, ai);
      }
      const __bf16* vtile = vbuf + (size_t)tm * 16 * HP;
#pragma unroll
      for (int k0 = 0; k0 < HP; k0 += 32) {
        v16bf af = load_a_frag(vtile, HP, k0, lane);
        v16bf bf_ = load_b_frag(Whhb, HP, tn * 16, k0, lane);
        ah = wmma_bf16(af, bf_, ah);
      }
      int col = tn * 16 + m;
#pragma unroll
      for (int r = 0; r < 8; ++r) {
        int rr = tm * 16 + half * 8 + r;
        float gi = ai[r] + bihp[col];
        float gh = ah[r] + bhhp[col];
        if (col < 2 * HP) {
          grz[rr * (2 * HP) + col] = sigmoidf_(gi + gh);   // r and z gates
        } else {
          inb[rr * HP + col - 2 * HP] = gi;                // i_n
          hnb[rr * HP + col - 2 * HP] = gh;                // h_n
        }
      }
    }
    __syncthreads();

    // ---- phase 6: GRU elementwise, state update, output ----
    for (int p = tid; p < BB * HH; p += 1024) {
      int b = p / HH, h = p % HH;
      float rg = grz[b * (2 * HP) + h];
      float zg = grz[b * (2 * HP) + HP + h];
      float nn = tanhf(inb[b * HP + h] + rg * hnb[b * HP + h]);
      float vo = vf32[b * HP + h];
      float vn = (1.0f - zg) * nn + zg * vo;
      vf32[b * HP + h] = vn;
      vbuf[b * HP + h] = (__bf16)vn;
      out[((size_t)t * BB + b) * HH + h] = vn;
    }
    __syncthreads();
  }
}

// ------------------------------- launcher ----------------------------------
extern "C" void kernel_launch(void* const* d_in, const int* in_sizes, int n_in,
                              void* d_out, int out_size, void* d_ws, size_t ws_size,
                              hipStream_t stream) {
  const float* Up   = (const float*)d_in[0];
  const float* Uq   = (const float*)d_in[1];
  const float* Wp   = (const float*)d_in[2];
  const float* Wq   = (const float*)d_in[3];
  const float* Wv   = (const float*)d_in[4];
  const float* Wg   = (const float*)d_in[5];
  const float* V    = (const float*)d_in[6];
  const float* v0   = (const float*)d_in[7];
  const float* W_ih = (const float*)d_in[8];
  const float* W_hh = (const float*)d_in[9];
  const float* b_ih = (const float*)d_in[10];
  const float* b_hh = (const float*)d_in[11];
  float* out = (float*)d_out;

  char*  ws  = (char*)d_ws;
  size_t off = 0;
  auto alloc = [&](size_t bytes) -> void* {
    void* p = ws + off;
    off = (off + bytes + 255) & ~(size_t)255;
    return p;
  };
  __bf16* Upbf   = (__bf16*)alloc((size_t)LP * BB * DD * 2);
  __bf16* Uqbf   = (__bf16*)alloc((size_t)LQ * BB * DD * 2);
  float*  WupAll = (float*) alloc((size_t)LP * BB * HP * 4);
  float*  WuqAll = (float*) alloc((size_t)LQ * BB * HP * 4);
  __bf16* Wpc    = (__bf16*)alloc((size_t)HP * DD * 2);
  __bf16* Wqc    = (__bf16*)alloc((size_t)HP * DD * 2);
  __bf16* Wgc    = (__bf16*)alloc((size_t)512 * 1024 * 2);
  __bf16* Wihb   = (__bf16*)alloc((size_t)3 * HP * 512 * 2);
  __bf16* Whhb   = (__bf16*)alloc((size_t)3 * HP * HP * 2);
  __bf16* Wvb    = (__bf16*)alloc((size_t)HP * HP * 2);
  float*  bihp   = (float*) alloc((size_t)3 * HP * 4);
  float*  bhhp   = (float*) alloc((size_t)3 * HP * 4);

  // prep: bf16 conversions + padded/combined weights
  {
    int n = LP * BB * DD;
    k_f32_to_bf16<<<(n + 255) / 256, 256, 0, stream>>>(Up, Upbf, n);
    n = LQ * BB * DD;
    k_f32_to_bf16<<<(n + 255) / 256, 256, 0, stream>>>(Uq, Uqbf, n);
  }
  k_build_wc <<<(HP * DD + 255) / 256, 256, 0, stream>>>(Wp, Wpc);
  k_build_wc <<<(HP * DD + 255) / 256, 256, 0, stream>>>(Wq, Wqc);
  k_build_wg <<<(512 * 1024 + 255) / 256, 256, 0, stream>>>(Wg, Wgc);
  k_build_wih<<<(3 * HP * 512 + 255) / 256, 256, 0, stream>>>(W_ih, Wihb);
  k_build_whh<<<(3 * HP * HP + 255) / 256, 256, 0, stream>>>(W_hh, Whhb);
  k_build_wv <<<(HP * HP + 255) / 256, 256, 0, stream>>>(Wv, Wvb);
  k_build_bias<<<(3 * HP + 255) / 256, 256, 0, stream>>>(b_ih, bihp);
  k_build_bias<<<(3 * HP + 255) / 256, 256, 0, stream>>>(b_hh, bhhp);

  // hoisted projections (parallel WMMA GEMMs)
  {
    int mtiles = (LP * BB) / 16;                       // 2000
    int tiles  = mtiles * (HP / 16);                   // 20000
    k_proj_gemm<<<(tiles + 15) / 16, 512, 0, stream>>>(Upbf, Wpc, WupAll, mtiles);
    mtiles = (LQ * BB) / 16;                           // 256
    tiles  = mtiles * (HP / 16);
    k_proj_gemm<<<(tiles + 15) / 16, 512, 0, stream>>>(Uqbf, Wqc, WuqAll, mtiles);
  }

  // sequential scan: one persistent workgroup, 284KB dynamic LDS
  hipFuncSetAttribute((const void*)k_seq_scan,
                      hipFuncAttributeMaxDynamicSharedMemorySize, SMEM_BYTES);
  k_seq_scan<<<1, 1024, SMEM_BYTES, stream>>>(
      Upbf, WupAll, WuqAll, Uq, V, Wgc, Wihb, Whhb, Wvb, bihp, bhhp, v0, out);
}
